// BNN_VGG7_TLUTRAIN_32461362823714
// MI455X (gfx1250) — 8-bit matrix operand order repaired and checked by reading, compiled, not run on hardware
//
#include <hip/hip_runtime.h>

typedef signed char i8;
typedef __attribute__((ext_vector_type(8))) int v8i;

#define EPSV 1e-5f

__global__ void bn_scale_kernel(const float* __restrict__ g, const float* __restrict__ v,
                                float* __restrict__ s, int C) {
  int i = blockIdx.x * 256 + threadIdx.x;
  if (i < C) s[i] = g[i] / sqrtf(v[i] + EPSV);
}

__global__ void pack_conv_kernel(const float* __restrict__ w, i8* __restrict__ out,
                                 int Cin, int Cout, int total) {
  int idx = blockIdx.x * 256 + threadIdx.x;
  if (idx >= total) return;
  int b    = idx & 3;
  int vg   = (idx >> 2) & 7;
  int lane = (idx >> 5) & 31;
  int blk  = idx >> 10;
  int KC = Cin >> 6, CoutTiles = Cout >> 4;
  int kc  = blk % KC;
  int t2  = blk / KC;
  int cot = t2 % CoutTiles;
  int tap = t2 / CoutTiles;
  int N = lane & 15;
  int K = ((vg >> 1) << 4) + ((lane < 16) ? 0 : 8) + ((vg & 1) << 2) + b;
  int co = (cot << 4) + N;
  int ci = (kc << 6) + K;
  float wv = w[(size_t)(co * Cin + ci) * 9 + tap];
  out[idx] = (wv >= 0.f) ? (i8)1 : (i8)-1;
}

__global__ void pack_fc1_kernel(const float* __restrict__ w, i8* __restrict__ out, int total) {
  int idx = blockIdx.x * 256 + threadIdx.x;
  if (idx >= total) return;
  int b    = idx & 3;
  int vg   = (idx >> 2) & 7;
  int lane = (idx >> 5) & 31;
  int blk  = idx >> 10;
  int kc    = blk & 127;
  int ntile = blk >> 7;
  int N = lane & 15;
  int K = ((vg >> 1) << 4) + ((lane < 16) ? 0 : 8) + ((vg & 1) << 2) + b;
  int o = (ntile << 4) + N;
  int k = (kc << 6) + K;
  int refk = (k & 511) * 16 + (k >> 9);
  float wv = w[(size_t)o * 8192 + refk];
  out[idx] = (wv >= 0.f) ? (i8)1 : (i8)-1;
}

__global__ void pack_fc2_kernel(const float* __restrict__ w, i8* __restrict__ out, int total) {
  int idx = blockIdx.x * 256 + threadIdx.x;
  if (idx >= total) return;
  out[idx] = (w[idx] >= 0.f) ? (i8)1 : (i8)-1;
}

__global__ void conv0_kernel(const float* __restrict__ x, const float* __restrict__ w,
                             const float* __restrict__ sA, const float* __restrict__ bA,
                             const float* __restrict__ mA, i8* __restrict__ out) {
  int co = threadIdx.x & 127;
  int p  = blockIdx.x * 2 + (threadIdx.x >> 7);
  int n  = p >> 10, rem = p & 1023, y = rem >> 5, xx = rem & 31;
  float acc = 0.f;
  #pragma unroll
  for (int ci = 0; ci < 3; ++ci)
    #pragma unroll
    for (int dy = -1; dy <= 1; ++dy) {
      int yc = y + dy; yc = yc < 0 ? 0 : (yc > 31 ? 31 : yc);
      #pragma unroll
      for (int dx = -1; dx <= 1; ++dx) {
        int xc = xx + dx; xc = xc < 0 ? 0 : (xc > 31 ? 31 : xc);
        float xv = x[((n * 3 + ci) * 32 + yc) * 32 + xc];
        float wv = w[((co * 3 + ci) * 3 + (dy + 1)) * 3 + (dx + 1)];
        acc += (wv >= 0.f) ? xv : -xv;
      }
    }
  float vbn = (acc - mA[co]) * sA[co] + bA[co];
  out[(size_t)p * 128 + co] = (vbn >= 0.f) ? (i8)1 : (i8)-1;
}

template <int CIN, int COUT, int H, int W>
__global__ void bconv_wmma_kernel(const i8* __restrict__ in, const i8* __restrict__ wp,
                                  const float* __restrict__ sA, const float* __restrict__ bA,
                                  const float* __restrict__ mA, i8* __restrict__ out) {
  constexpr int KC = CIN >> 6;
  constexpr int CoutTiles = COUT >> 4;
  const int lane = threadIdx.x & 31;
  const int wave = threadIdx.x >> 5;
  const int cot = blockIdx.y * 8 + wave;
  if (cot >= CoutTiles) return;

  const int p0 = blockIdx.x << 4;
  const int m  = lane & 15;
  const int hi = lane >> 4;
  const int p  = p0 + m;
  const int n  = p / (H * W);
  const int rem = p - n * (H * W);
  const int y = rem / W;
  const int x = rem - y * W;

  const int co = (cot << 4) + m;
  const float sc = sA[co], bc = bA[co], mc = mA[co];

  v8i acc = {};
  #pragma unroll
  for (int tap = 0; tap < 9; ++tap) {
    const int dy = tap / 3 - 1, dx = tap % 3 - 1;
    int yc = y + dy; yc = yc < 0 ? 0 : (yc >= H ? H - 1 : yc);
    int xc = x + dx; xc = xc < 0 ? 0 : (xc >= W ? W - 1 : xc);
    const i8* arow = in + (size_t)((n * H + yc) * W + xc) * CIN + (hi << 3);
    const i8* wrow = wp + (size_t)((tap * CoutTiles + cot) * KC) * 1024 + lane * 32;
    #pragma unroll
    for (int kc = 0; kc < KC; ++kc) {
      const i8* ap = arow + (kc << 6);
      int2 d0 = *(const int2*)(ap);
      int2 d1 = *(const int2*)(ap + 16);
      int2 d2 = *(const int2*)(ap + 32);
      int2 d3 = *(const int2*)(ap + 48);
      v8i A; A[0]=d0.x; A[1]=d0.y; A[2]=d1.x; A[3]=d1.y;
             A[4]=d2.x; A[5]=d2.y; A[6]=d3.x; A[7]=d3.y;
      const int4 w0 = *(const int4*)(wrow + kc * 1024);
      const int4 w1 = *(const int4*)(wrow + kc * 1024 + 16);
      v8i B; B[0]=w0.x; B[1]=w0.y; B[2]=w0.z; B[3]=w0.w;
             B[4]=w1.x; B[5]=w1.y; B[6]=w1.z; B[7]=w1.w;
      acc = __builtin_amdgcn_wmma_i32_16x16x64_iu8(true, A, true, B, acc, false, false);
    }
  }
  #pragma unroll
  for (int r = 0; r < 8; ++r) {
    int pp = p0 + r + (hi << 3);
    float vbn = ((float)acc[r] - mc) * sc + bc;
    out[(size_t)pp * COUT + co] = (vbn >= 0.f) ? (i8)1 : (i8)-1;
  }
}

__global__ void pool2_kernel(const i8* __restrict__ in, i8* __restrict__ out,
                             int H, int W, int C, int total) {
  int idx = blockIdx.x * 256 + threadIdx.x;
  if (idx >= total) return;
  int c = idx % C;
  int r = idx / C;
  int Wh = W >> 1, Hh = H >> 1;
  int xo = r % Wh; r /= Wh;
  int yo = r % Hh;
  int n  = r / Hh;
  const i8* base = in + (size_t)((n * H + yo * 2) * W + xo * 2) * C + c;
  i8 a = base[0], b = base[C], cc = base[(size_t)W * C], d = base[(size_t)W * C + C];
  i8 m0 = a > b ? a : b;
  i8 m1 = cc > d ? cc : d;
  out[idx] = m0 > m1 ? m0 : m1;
}

__global__ void fc1_wmma_kernel(const i8* __restrict__ in, const i8* __restrict__ wp,
                                const float* __restrict__ sA, const float* __restrict__ bA,
                                const float* __restrict__ mA, i8* __restrict__ out) {
  const int lane  = threadIdx.x & 31;
  const int wave  = threadIdx.x >> 5;
  const int ntile = blockIdx.y * 8 + wave;
  const int mtile = blockIdx.x;
  const int m  = lane & 15;
  const int hi = lane >> 4;
  const int row = (mtile << 4) + m;
  const int co  = (ntile << 4) + m;
  const float sc = sA[co], bc = bA[co], mc = mA[co];

  const i8* arow = in + (size_t)row * 8192 + (hi << 3);
  const i8* wbase = wp + (size_t)(ntile * 128) * 1024 + lane * 32;

  v8i acc = {};
  #pragma unroll 8
  for (int kc = 0; kc < 128; ++kc) {
    const i8* ap = arow + (kc << 6);
    int2 d0 = *(const int2*)(ap);
    int2 d1 = *(const int2*)(ap + 16);
    int2 d2 = *(const int2*)(ap + 32);
    int2 d3 = *(const int2*)(ap + 48);
    v8i A; A[0]=d0.x; A[1]=d0.y; A[2]=d1.x; A[3]=d1.y;
           A[4]=d2.x; A[5]=d2.y; A[6]=d3.x; A[7]=d3.y;
    const int4 w0 = *(const int4*)(wbase + (size_t)kc * 1024);
    const int4 w1 = *(const int4*)(wbase + (size_t)kc * 1024 + 16);
    v8i B; B[0]=w0.x; B[1]=w0.y; B[2]=w0.z; B[3]=w0.w;
           B[4]=w1.x; B[5]=w1.y; B[6]=w1.z; B[7]=w1.w;
    acc = __builtin_amdgcn_wmma_i32_16x16x64_iu8(true, A, true, B, acc, false, false);
  }
  #pragma unroll
  for (int r = 0; r < 8; ++r) {
    int rr = (mtile << 4) + r + (hi << 3);
    float vbn = ((float)acc[r] - mc) * sc + bc;
    out[(size_t)rr * 1024 + co] = (vbn >= 0.f) ? (i8)1 : (i8)-1;
  }
}

__global__ void fc2_kernel(const i8* __restrict__ h, const i8* __restrict__ w,
                           const float* __restrict__ scale, float* __restrict__ out) {
  int idx = blockIdx.x * 256 + threadIdx.x;
  if (idx >= 2560) return;
  int n = idx / 10, o = idx % 10;
  const i8* a = h + (size_t)n * 1024;
  const i8* b = w + (size_t)o * 1024;
  int acc = 0;
  #pragma unroll 8
  for (int k = 0; k < 1024; ++k) acc += (int)a[k] * (int)b[k];
  out[idx] = (float)acc * scale[0];
}

extern "C" void kernel_launch(void* const* d_in, const int* in_sizes, int n_in,
                              void* d_out, int out_size, void* d_ws, size_t ws_size,
                              hipStream_t stream) {
  (void)in_sizes; (void)n_in; (void)out_size; (void)ws_size;
  const float* x  = (const float*)d_in[0];
  const float* cw[6]; for (int i = 0; i < 6; ++i) cw[i] = (const float*)d_in[1 + i];
  const float* fw1 = (const float*)d_in[7];
  const float* fw2 = (const float*)d_in[8];
  const float *bg[7], *bb[7], *bm[7], *bv[7];
  for (int L = 0; L < 7; ++L) {
    bg[L] = (const float*)d_in[9 + 4 * L + 0];
    bb[L] = (const float*)d_in[9 + 4 * L + 1];
    bm[L] = (const float*)d_in[9 + 4 * L + 2];
    bv[L] = (const float*)d_in[9 + 4 * L + 3];
  }
  const float* scale = (const float*)d_in[37];

  char* ws = (char*)d_ws;
  size_t off = 0;
  auto alloc = [&](size_t bytes) { size_t o = off; off = (off + bytes + 255) & ~(size_t)255; return o; };
  const int bnC[7] = {128, 128, 256, 256, 512, 512, 1024};
  float* sArr[7];
  for (int L = 0; L < 7; ++L) sArr[L] = (float*)(ws + alloc((size_t)bnC[L] * 4));
  const int packTot[5] = {9*8*2*1024, 9*16*2*1024, 9*16*4*1024, 9*32*4*1024, 9*32*8*1024};
  i8* wpk[5];
  for (int L = 0; L < 5; ++L) wpk[L] = (i8*)(ws + alloc((size_t)packTot[L]));
  i8* fpk1 = (i8*)(ws + alloc(64 * 128 * 1024));
  i8* fpk2 = (i8*)(ws + alloc(10 * 1024));
  i8* bufX = (i8*)(ws + alloc(33554432));
  i8* bufY = (i8*)(ws + alloc(33554432));

  for (int L = 0; L < 7; ++L)
    bn_scale_kernel<<<dim3((bnC[L] + 255) / 256), dim3(256), 0, stream>>>(bg[L], bv[L], sArr[L], bnC[L]);

  const int cinL[5]  = {128, 128, 256, 256, 512};
  const int coutL[5] = {128, 256, 256, 512, 512};
  for (int L = 0; L < 5; ++L)
    pack_conv_kernel<<<dim3((packTot[L] + 255) / 256), dim3(256), 0, stream>>>(
        cw[L + 1], wpk[L], cinL[L], coutL[L], packTot[L]);
  pack_fc1_kernel<<<dim3(64 * 128 * 1024 / 256), dim3(256), 0, stream>>>(fw1, fpk1, 64 * 128 * 1024);
  pack_fc2_kernel<<<dim3(10240 / 256), dim3(256), 0, stream>>>(fw2, fpk2, 10240);

  conv0_kernel<<<dim3(256 * 1024 / 2), dim3(256), 0, stream>>>(x, cw[0], sArr[0], bb[0], bm[0], bufX);

  bconv_wmma_kernel<128, 128, 32, 32><<<dim3(16384, 1), dim3(256), 0, stream>>>(
      bufX, wpk[0], sArr[1], bb[1], bm[1], bufY);
  pool2_kernel<<<dim3(8388608 / 256), dim3(256), 0, stream>>>(bufY, bufX, 32, 32, 128, 8388608);

  bconv_wmma_kernel<128, 256, 16, 16><<<dim3(4096, 2), dim3(256), 0, stream>>>(
      bufX, wpk[1], sArr[2], bb[2], bm[2], bufY);
  bconv_wmma_kernel<256, 256, 16, 16><<<dim3(4096, 2), dim3(256), 0, stream>>>(
      bufY, wpk[2], sArr[3], bb[3], bm[3], bufX);
  pool2_kernel<<<dim3(4194304 / 256), dim3(256), 0, stream>>>(bufX, bufY, 16, 16, 256, 4194304);

  bconv_wmma_kernel<256, 512, 8, 8><<<dim3(1024, 4), dim3(256), 0, stream>>>(
      bufY, wpk[3], sArr[4], bb[4], bm[4], bufX);
  bconv_wmma_kernel<512, 512, 8, 8><<<dim3(1024, 4), dim3(256), 0, stream>>>(
      bufX, wpk[4], sArr[5], bb[5], bm[5], bufY);
  pool2_kernel<<<dim3(2097152 / 256), dim3(256), 0, stream>>>(bufY, bufX, 8, 8, 512, 2097152);

  fc1_wmma_kernel<<<dim3(16, 8), dim3(256), 0, stream>>>(bufX, fpk1, sArr[6], bb[6], bm[6], bufY);

  fc2_kernel<<<dim3(10), dim3(256), 0, stream>>>(bufY, fpk2, scale, (float*)d_out);
}
